// LlamaDecoderLayer_57878979281333
// MI455X (gfx1250) — matrix operand order repaired and checked by reading, not compiled, not run on hardware
//
#include <hip/hip_runtime.h>

// ---------------------------------------------------------------------------
// Llama decoder layer for gfx1250 (MI455X).
// Compute-bound (AI ~1000 FLOP/B) -> v_wmma_f32_16x16x32_bf16 everywhere,
// fp32 accumulate. Weights converted fp32->bf16 on the fly in the GEMM's
// cooperative LDS load; bf16 A-tiles copied with GLOBAL_LOAD_ASYNC_TO_LDS_B128
// (ASYNCcnt) to overlap with the B conversion. Flash attention: 4 waves per
// block, 1 wave per (head, 16-row query tile), online softmax, P staged
// through wave-private LDS (same-wave DS ordering, no barriers).
// ---------------------------------------------------------------------------

#define S_LEN 2048
#define H_DIM 4096
#define NHEAD 32
#define NKVH 8
#define HEADD 128
#define I_DIM 11008

typedef __attribute__((ext_vector_type(16))) __bf16 v16bf;
typedef __attribute__((ext_vector_type(8)))  float  v8f;
typedef unsigned int u32x4 __attribute__((ext_vector_type(4)));
typedef float        f32x4 __attribute__((ext_vector_type(4)));

union FragU { u32x4 q[2]; v16bf v; };

__device__ __forceinline__ unsigned short f2bf(float f) {
  union { float f; unsigned u; } c; c.f = f;
  unsigned u = c.u;
  u += 0x7FFFu + ((u >> 16) & 1u);          // round-to-nearest-even
  return (unsigned short)(u >> 16);
}

// 16 contiguous bf16 (B-operand fragments, contiguous-K layouts)
__device__ __forceinline__ v16bf ld_frag16(const unsigned short* p) {
  FragU u;
  u.q[0] = *(const u32x4*)(p);
  u.q[1] = *(const u32x4*)(p + 8);
  return u.v;
}
// two 8-element chunks (A-operand: K {kb..kb+7} then {kb+8..kb+15})
__device__ __forceinline__ v16bf ld_frag8x2(const unsigned short* p0,
                                            const unsigned short* p1) {
  FragU u;
  u.q[0] = *(const u32x4*)(p0);
  u.q[1] = *(const u32x4*)(p1);
  return u.v;
}

__device__ __forceinline__ v8f wmma_bf16(v16bf a, v16bf b, v8f c) {
  return __builtin_amdgcn_wmma_f32_16x16x32_bf16(false, a, false, b,
                                                 (short)0, c, false, false);
}

// gfx1250 async memory->LDS copy (tracked with ASYNCcnt).
// Generic pointers to LDS carry the DS byte offset in their low 32 bits
// (aperture maps generic->LDS by truncation), so we can feed them directly.
__device__ __forceinline__ void async_copy_b128(void* lds, const void* g) {
  const unsigned lds_off = (unsigned)(size_t)lds;
  const unsigned long long ga = (unsigned long long)(size_t)g;
  asm volatile("global_load_async_to_lds_b128 %0, %1, off"
               :: "v"(lds_off), "v"(ga)
               : "memory");
}
__device__ __forceinline__ void wait_async0() {
  asm volatile("s_wait_asynccnt 0x0" ::: "memory");
}

// ---------------------------------------------------------------------------
// RMSNorm: fp32 [rows, H_DIM] -> bf16, one 256-thread block per row
// ---------------------------------------------------------------------------
__global__ __launch_bounds__(256) void rmsnorm_kernel(
    const float* __restrict__ x, const float* __restrict__ w,
    unsigned short* __restrict__ out) {
  const int row = blockIdx.x;
  const float* xr = x + (size_t)row * H_DIM;
  float vals[16];
  float ss = 0.f;
#pragma unroll
  for (int i = 0; i < 4; ++i) {
    const int idx = i * 1024 + threadIdx.x * 4;
    f32x4 f = *(const f32x4*)(xr + idx);
#pragma unroll
    for (int j = 0; j < 4; ++j) { vals[i * 4 + j] = f[j]; ss += f[j] * f[j]; }
  }
#pragma unroll
  for (int d = 1; d < 32; d <<= 1) ss += __shfl_xor(ss, d, 32);
  __shared__ float red[8];
  if ((threadIdx.x & 31) == 0) red[threadIdx.x >> 5] = ss;
  __syncthreads();
  float tot = 0.f;
#pragma unroll
  for (int i = 0; i < 8; ++i) tot += red[i];
  const float r = rsqrtf(tot * (1.0f / H_DIM) + 1e-5f);
  unsigned short* orow = out + (size_t)row * H_DIM;
#pragma unroll
  for (int i = 0; i < 4; ++i) {
    const int idx = i * 1024 + threadIdx.x * 4;
#pragma unroll
    for (int j = 0; j < 4; ++j)
      orow[idx + j] = f2bf(vals[i * 4 + j] * r * w[idx + j]);
  }
}

// ---------------------------------------------------------------------------
// WMMA GEMM: C[M,N] = A[M,K](bf16) @ B[K,N](fp32, ldb row stride)
// 128x128 tile per 256-thread block; 8 waves of 64x32; BK=32.
// A tile arrives via async LDS DMA; B tile converted fp32->bf16 in flight.
// MODE 0: Cf = A@B              (fp32 out)
// MODE 1: Cf = res + A@B        (fp32 out)
// MODE 2: Cb = silu(A@Bg)*A@Bu  (bf16 out; Bu at column offset n2off)
// ---------------------------------------------------------------------------
template <int MODE>
__global__ __launch_bounds__(256) void gemm_wmma_kernel(
    const unsigned short* __restrict__ A, const float* __restrict__ B,
    const float* __restrict__ res, float* __restrict__ Cf,
    unsigned short* __restrict__ Cb, int M, int N, int K, int ldb, int n2off) {
  __shared__ unsigned short As[128][32];             // [m][k]
  __shared__ unsigned short Bst[128][32];            // transposed: [n][k]
  __shared__ unsigned short Bst2[(MODE == 2) ? 128 : 1][(MODE == 2) ? 32 : 1];

  const int tid = threadIdx.x;
  const int L = tid & 31;
  const int w = tid >> 5;
  const int wRow = w >> 2;        // 0..1 -> 64 rows
  const int wCol = w & 3;         // 0..3 -> 32 cols
  const int m0 = blockIdx.y * 128;
  const int n0 = blockIdx.x * 128;
  const int lane16 = L & 15;
  const int half = L >> 4;

  v8f acc[4][2] = {};
  v8f acc2[4][2] = {};

  for (int k0 = 0; k0 < K; k0 += 32) {
    // A tile: 128x32 bf16 = 8KB, async DMA to LDS, 2 x B128 per thread
#pragma unroll
    for (int i = 0; i < 2; ++i) {
      const int c = tid * 2 + i;         // 0..511 16B chunks
      const int row = c >> 2;
      const int col8 = (c & 3) * 8;
      async_copy_b128(&As[row][col8],
                      A + (size_t)(m0 + row) * K + k0 + col8);
    }
    // B tile: 32x128 fp32 -> bf16, stored transposed; 4 float4 per thread
#pragma unroll
    for (int i = 0; i < 4; ++i) {
      const int u = tid + i * 256;
      const int row = u >> 5;            // k within tile
      const int c4 = (u & 31) << 2;      // n within tile
      f32x4 f = *(const f32x4*)(B + (size_t)(k0 + row) * ldb + n0 + c4);
#pragma unroll
      for (int j = 0; j < 4; ++j) Bst[c4 + j][row] = f2bf(f[j]);
      if constexpr (MODE == 2) {
        f32x4 g =
            *(const f32x4*)(B + (size_t)(k0 + row) * ldb + n2off + n0 + c4);
#pragma unroll
        for (int j = 0; j < 4; ++j) Bst2[c4 + j][row] = f2bf(g[j]);
      }
      if (i == 0 && k0 + 32 < K)         // hint next B tile toward L2
        __builtin_prefetch(B + (size_t)(k0 + 32 + row) * ldb + n0 + c4, 0, 1);
    }
    wait_async0();
    __syncthreads();

    const int kbA = half * 16;
    v16bf af[4];
#pragma unroll
    for (int i = 0; i < 4; ++i) {
      const int r = wRow * 64 + i * 16 + lane16;
      af[i] = ld_frag8x2(&As[r][kbA], &As[r][kbA + 8]);
    }
    const int kbB = half * 16;
#pragma unroll
    for (int j = 0; j < 2; ++j) {
      const int c = wCol * 32 + j * 16 + lane16;
      v16bf bf = ld_frag16(&Bst[c][kbB]);
#pragma unroll
      for (int i = 0; i < 4; ++i) acc[i][j] = wmma_bf16(af[i], bf, acc[i][j]);
      if constexpr (MODE == 2) {
        v16bf b2 = ld_frag16(&Bst2[c][kbB]);
#pragma unroll
        for (int i = 0; i < 4; ++i)
          acc2[i][j] = wmma_bf16(af[i], b2, acc2[i][j]);
      }
    }
    __syncthreads();
  }

  // epilogue: C layout -> VGPR r holds rows r (lanes 0-15) / r+8 (lanes 16-31)
#pragma unroll
  for (int i = 0; i < 4; ++i) {
#pragma unroll
    for (int j = 0; j < 2; ++j) {
#pragma unroll
      for (int r = 0; r < 8; ++r) {
        const int m = m0 + wRow * 64 + i * 16 + r + half * 8;
        const int n = n0 + wCol * 32 + j * 16 + lane16;
        const float v = acc[i][j][r];
        if constexpr (MODE == 0) {
          Cf[(size_t)m * N + n] = v;
        } else if constexpr (MODE == 1) {
          Cf[(size_t)m * N + n] = res[(size_t)m * N + n] + v;
        } else {
          const float g = v;
          const float up = acc2[i][j][r];
          const float s = g / (1.f + __expf(-g));
          Cb[(size_t)m * N + n] = f2bf(s * up);
        }
      }
    }
  }
}

// ---------------------------------------------------------------------------
// RoPE on q and k heads of qkv (fp32) -> bf16 Q [S,4096], K [S,1024]
// ---------------------------------------------------------------------------
__global__ __launch_bounds__(256) void rope_qk_kernel(
    const int* __restrict__ pos, const float* __restrict__ qkv,
    unsigned short* __restrict__ Q, unsigned short* __restrict__ Kc) {
  const int idx = blockIdx.x * blockDim.x + threadIdx.x;
  const int i = idx & 63;
  const int h = (idx >> 6) % 40;
  const int s = idx / (40 * 64);
  const float p = (float)pos[s];
  const float inv = __powf(10000.f, -(float)i * (1.f / 64.f));
  float sn, cs;
  __sincosf(p * inv, &sn, &cs);
  const float* base = qkv + (size_t)s * 6144 + h * HEADD;
  const float x1 = base[i];
  const float x2 = base[i + 64];
  const float o1 = x1 * cs - x2 * sn;
  const float o2 = x2 * cs + x1 * sn;
  if (h < NHEAD) {
    unsigned short* o = Q + (size_t)s * (NHEAD * HEADD) + h * HEADD;
    o[i] = f2bf(o1);
    o[i + 64] = f2bf(o2);
  } else {
    unsigned short* o = Kc + (size_t)s * (NKVH * HEADD) + (h - NHEAD) * HEADD;
    o[i] = f2bf(o1);
    o[i + 64] = f2bf(o2);
  }
}

// v part of qkv -> transposed bf16 VT [NKVH*HEADD, S]
__global__ __launch_bounds__(256) void vtrans_kernel(
    const float* __restrict__ qkv, unsigned short* __restrict__ VT) {
  const int idx = blockIdx.x * blockDim.x + threadIdx.x;
  const int d = idx & (NKVH * HEADD - 1);
  const int s = idx >> 10;
  VT[(size_t)d * S_LEN + s] =
      f2bf(qkv[(size_t)s * 6144 + (NHEAD * HEADD) + d]);
}

// ---------------------------------------------------------------------------
// Flash attention: 128-thread blocks = 4 independent waves; each wave owns
// one (16-query tile, head) pair. Causal, GQA rep=4. No barriers: P is
// wave-private LDS and same-wave DS ops are ordered (DScnt, in order).
// ---------------------------------------------------------------------------
__global__ __launch_bounds__(128) void flash_attn_kernel(
    const unsigned short* __restrict__ Q, const unsigned short* __restrict__ Kc,
    const unsigned short* __restrict__ VT, unsigned short* __restrict__ O) {
  __shared__ unsigned short P[4][16][32];
  const int L = threadIdx.x & 31;
  const int wv = threadIdx.x >> 5;
  const int q0 = (blockIdx.x * 4 + wv) * 16;
  const int h = blockIdx.y;
  const int hk = h >> 2;
  const int lane16 = L & 15;
  const int half = L >> 4;
  const int kbA = half * 8;    // unused; A-frags use kbB too
  const int kbB = half * 16;   // B-frag K base
  const int rbase = half * 8;  // C-frag row base

  v16bf qf[4];
  {
    const unsigned short* qrow =
        Q + (size_t)(q0 + lane16) * (NHEAD * HEADD) + h * HEADD;
#pragma unroll
    for (int dk = 0; dk < 4; ++dk)
      qf[dk] = ld_frag8x2(qrow + dk * 32 + kbB, qrow + dk * 32 + 8 + kbB);
  }

  v8f acc[8] = {};
  float mI[8], lI[8];
#pragma unroll
  for (int r = 0; r < 8; ++r) { mI[r] = -1e30f; lI[r] = 0.f; }

  const float scale = 0.088388347648318447f;  // 1/sqrt(128)
  const int kend = q0 + 16;
  for (int kb = 0; kb < kend; kb += 32) {
    v8f sc[2] = {};
#pragma unroll
    for (int t = 0; t < 2; ++t) {
      const unsigned short* krow =
          Kc + (size_t)(kb + t * 16 + lane16) * (NKVH * HEADD) + hk * HEADD;
#pragma unroll
      for (int dk = 0; dk < 4; ++dk)
        sc[t] = wmma_bf16(qf[dk], ld_frag16(krow + dk * 32 + kbB), sc[t]);
    }
    float corr[8];
#pragma unroll
    for (int r = 0; r < 8; ++r) {
      const int qg = q0 + rbase + r;
      float s0 = sc[0][r] * scale;
      float s1 = sc[1][r] * scale;
      if (kb + lane16 > qg) s0 = -1e30f;
      if (kb + 16 + lane16 > qg) s1 = -1e30f;
      float mx = fmaxf(s0, s1);
#pragma unroll
      for (int d = 1; d < 16; d <<= 1) mx = fmaxf(mx, __shfl_xor(mx, d, 32));
      mx = fmaxf(mx, mI[r]);
      const float p0 = __expf(s0 - mx);
      const float p1 = __expf(s1 - mx);
      float sum = p0 + p1;
#pragma unroll
      for (int d = 1; d < 16; d <<= 1) sum += __shfl_xor(sum, d, 32);
      corr[r] = __expf(mI[r] - mx);
      lI[r] = lI[r] * corr[r] + sum;
      mI[r] = mx;
      P[wv][rbase + r][lane16] = f2bf(p0);
      P[wv][rbase + r][16 + lane16] = f2bf(p1);
    }
#pragma unroll
    for (int j = 0; j < 8; ++j)
#pragma unroll
      for (int r = 0; r < 8; ++r) acc[j][r] *= corr[r];
    const v16bf pf =
        ld_frag8x2(&P[wv][lane16][kbB], &P[wv][lane16][8 + kbB]);
#pragma unroll
    for (int j = 0; j < 8; ++j) {
      const unsigned short* vrow =
          VT + (size_t)(hk * HEADD + j * 16 + lane16) * S_LEN + kb + kbB;
      acc[j] = wmma_bf16(pf, ld_frag16(vrow), acc[j]);
    }
  }
#pragma unroll
  for (int j = 0; j < 8; ++j) {
#pragma unroll
    for (int r = 0; r < 8; ++r) {
      const int m = q0 + rbase + r;
      O[(size_t)m * (NHEAD * HEADD) + h * HEADD + j * 16 + lane16] =
          f2bf(acc[j][r] / lI[r]);
    }
  }
}

// ---------------------------------------------------------------------------
extern "C" void kernel_launch(void* const* d_in, const int* in_sizes, int n_in,
                              void* d_out, int out_size, void* d_ws,
                              size_t ws_size, hipStream_t stream) {
  (void)in_sizes; (void)n_in; (void)out_size; (void)ws_size;
  const int*   positions = (const int*)d_in[0];
  const float* hidden    = (const float*)d_in[1];
  const float* w_qkv     = (const float*)d_in[2];
  const float* w_o       = (const float*)d_in[3];
  const float* w_gate_up = (const float*)d_in[4];
  const float* w_down    = (const float*)d_in[5];
  const float* ln1       = (const float*)d_in[6];
  const float* ln2       = (const float*)d_in[7];
  float* out = (float*)d_out;

  size_t off = 0;
  auto alloc = [&](size_t bytes) {
    off = (off + 255) & ~(size_t)255;
    void* p = (char*)d_ws + off;
    off += bytes;
    return p;
  };
  unsigned short* xa  = (unsigned short*)alloc((size_t)S_LEN * H_DIM * 2);
  float*          qkv = (float*)alloc((size_t)S_LEN * 6144 * 4);
  unsigned short* Qb  = (unsigned short*)alloc((size_t)S_LEN * NHEAD * HEADD * 2);
  unsigned short* Kb  = (unsigned short*)alloc((size_t)S_LEN * NKVH * HEADD * 2);
  unsigned short* VT  = (unsigned short*)alloc((size_t)NKVH * HEADD * S_LEN * 2);
  unsigned short* Ob  = (unsigned short*)alloc((size_t)S_LEN * NHEAD * HEADD * 2);
  float*          h2  = (float*)alloc((size_t)S_LEN * H_DIM * 4);
  unsigned short* act = (unsigned short*)alloc((size_t)S_LEN * I_DIM * 2);

  // 1) ln1
  rmsnorm_kernel<<<S_LEN, 256, 0, stream>>>(hidden, ln1, xa);
  // 2) qkv = ln1(h) @ w_qkv   [2048 x 6144]
  gemm_wmma_kernel<0><<<dim3(6144 / 128, S_LEN / 128), 256, 0, stream>>>(
      xa, w_qkv, nullptr, qkv, nullptr, S_LEN, 6144, H_DIM, 6144, 0);
  // 3) rope + v transpose
  rope_qk_kernel<<<(S_LEN * 40 * 64) / 256, 256, 0, stream>>>(positions, qkv,
                                                              Qb, Kb);
  vtrans_kernel<<<(S_LEN * NKVH * HEADD) / 256, 256, 0, stream>>>(qkv, VT);
  // 4) flash attention (4 q-tiles per block)
  flash_attn_kernel<<<dim3(S_LEN / 64, NHEAD), 128, 0, stream>>>(Qb, Kb, VT,
                                                                 Ob);
  // 5) h2 = hidden + attn @ w_o
  gemm_wmma_kernel<1><<<dim3(H_DIM / 128, S_LEN / 128), 256, 0, stream>>>(
      Ob, w_o, hidden, h2, nullptr, S_LEN, H_DIM, H_DIM, H_DIM, 0);
  // 6) ln2
  rmsnorm_kernel<<<S_LEN, 256, 0, stream>>>(h2, ln2, xa);
  // 7) act = silu(x@Wg) * (x@Wu)   [2048 x 11008] bf16
  gemm_wmma_kernel<2><<<dim3(I_DIM / 128, S_LEN / 128), 256, 0, stream>>>(
      xa, w_gate_up, nullptr, nullptr, act, S_LEN, I_DIM, H_DIM, 2 * I_DIM,
      I_DIM);
  // 8) out = h2 + act @ w_down
  gemm_wmma_kernel<1><<<dim3(H_DIM / 128, S_LEN / 128), 256, 0, stream>>>(
      act, w_down, h2, out, nullptr, S_LEN, H_DIM, I_DIM, H_DIM, 0);
}
